// PointWiseDecoder2DSimple_5523327943115
// MI455X (gfx1250) — matrix operand order repaired and checked by reading, not compiled, not run on hardware
//
#include <hip/hip_runtime.h>
#include <math.h>

// ---------------------------------------------------------------------------
// Types for CDNA5 WMMA (wave32): v_wmma_f32_16x16x32_f16
// ---------------------------------------------------------------------------
typedef __attribute__((ext_vector_type(16))) _Float16 v16h;
typedef __attribute__((ext_vector_type(8)))  _Float16 v8h;
typedef __attribute__((ext_vector_type(8)))  float    v8f;

#define ROWS   32768   // b * nq = b * nc
#define BATCH  8
#define NQ     4096
#define NC     4096
#define LATENT 256
#define HEADS  4
#define DHEAD  256
#define INNER  1024

#define FLAG_GELU  1
#define FLAG_F32   2
#define FLAG_F16   4
#define FLAG_TF16  8

__device__ __forceinline__ float gelu_f(float x) {
  return 0.5f * x * (1.0f + erff(x * 0.70710678118654752440f));
}

union AF { v16h v; _Float16 e[16]; };

struct GemmP {
  const _Float16* A;    // [M][K] row-major, f16
  const _Float16* Bt;   // [N][K] row-major (i.e. B transposed), f16
  float*          outF;
  _Float16*       outH;
  const float*    bias; // [N] or null
  const float*    res;  // [M][ldr] or null
  int M, N, K;
  int lda, ldb, ldo, ldr;
  long long sAb, sAh, sBb, sBh, sOb, sOh; // per-z strides (z = zb*HZ + zh)
  int   HZ;
  float scale;
  int   flags;
};

// ---------------------------------------------------------------------------
// Generic WMMA GEMM: 8 waves/block, block tile 256(M) x 64(N), K step 32.
// Each wave owns a 32(M) x 64(N) strip: 2 A fragments x 4 B fragments = 8 acc.
// A fragment (16x32 f16): lane row = lane%16; halves 0..7 = K[k0+hi*8 ..],
//                         halves 8..15 = K[k0+16+hi*8 ..]   (contiguous 16B)
// B fragment (32x16 f16) from Bt[N][K]: lane col = lane%16;
//                         halves 0..7 = Bt[col][k0+hi*8 ..], 8..15 = Bt[col][k0+16+hi*8 ..]
// C/D (16x16 f32, 8 vgprs): row = r + 8*hi, col = lane%16.
// ---------------------------------------------------------------------------
__global__ __launch_bounds__(256) void gemm_wmma(GemmP p) {
  const int lane = threadIdx.x & 31;
  const int wave = threadIdx.x >> 5;
  const int hi   = lane >> 4;
  const int lc   = lane & 15;

  long long offA = 0, offB = 0, offO = 0;
  {
    const int z  = blockIdx.z;
    const int zb = z / p.HZ;
    const int zh = z - zb * p.HZ;
    offA = (long long)zb * p.sAb + (long long)zh * p.sAh;
    offB = (long long)zb * p.sBb + (long long)zh * p.sBh;
    offO = (long long)zb * p.sOb + (long long)zh * p.sOh;
  }
  const _Float16* A  = p.A  + offA;
  const _Float16* Bt = p.Bt + offB;

  const int m0 = blockIdx.y * 256 + wave * 32;
  const int n0 = blockIdx.x * 64;
  if (m0 >= p.M) return;

  const _Float16* aptr0 = A + (size_t)(m0 + lc) * p.lda;        // rows m0..m0+15
  const _Float16* aptr1 = A + (size_t)(m0 + 16 + lc) * p.lda;   // rows m0+16..m0+31

  v8f acc[2][4];
#pragma unroll
  for (int u = 0; u < 2; ++u)
#pragma unroll
    for (int t = 0; t < 4; ++t) acc[u][t] = (v8f)(0.0f);

  for (int k0 = 0; k0 < p.K; k0 += 32) {
    // Prefetch next K block of the A stream (global_prefetch_b8).
    __builtin_prefetch(aptr0 + k0 + 64, 0, 1);
    __builtin_prefetch(aptr1 + k0 + 64, 0, 1);

    AF a0, a1;
    *(v8h*)&a0.e[0] = *(const v8h*)(aptr0 + k0 + hi * 8);
    *(v8h*)&a0.e[8] = *(const v8h*)(aptr0 + k0 + 16 + hi * 8);
    *(v8h*)&a1.e[0] = *(const v8h*)(aptr1 + k0 + hi * 8);
    *(v8h*)&a1.e[8] = *(const v8h*)(aptr1 + k0 + 16 + hi * 8);
#pragma unroll
    for (int t = 0; t < 4; ++t) {
      AF b;
      const _Float16* bp = Bt + (size_t)(n0 + t * 16 + lc) * p.ldb + k0 + hi * 8;
      *(v8h*)&b.e[0] = *(const v8h*)(bp);
      *(v8h*)&b.e[8] = *(const v8h*)(bp + 16);
      acc[0][t] = __builtin_amdgcn_wmma_f32_16x16x32_f16(
          false, a0.v, false, b.v, (short)0, acc[0][t], false, false);
      acc[1][t] = __builtin_amdgcn_wmma_f32_16x16x32_f16(
          false, a1.v, false, b.v, (short)0, acc[1][t], false, false);
    }
  }

#pragma unroll
  for (int u = 0; u < 2; ++u) {
#pragma unroll
    for (int t = 0; t < 4; ++t) {
#pragma unroll
      for (int r = 0; r < 8; ++r) {
        const int orow = m0 + u * 16 + hi * 8 + r;
        const int ocol = n0 + t * 16 + lc;
        float v = acc[u][t][r] * p.scale;
        if (p.bias) v += p.bias[ocol];
        if (p.res)  v += p.res[(size_t)orow * p.ldr + ocol];
        if (p.flags & FLAG_GELU) v = gelu_f(v);
        if (p.flags & FLAG_F32)
          p.outF[offO + (size_t)orow * p.ldo + ocol] = v;
        if (p.flags & FLAG_F16)
          p.outH[offO + (size_t)orow * p.ldo + ocol] = (_Float16)v;
        if (p.flags & FLAG_TF16)  // transposed store (kv -> Bt layout for next GEMM)
          p.outH[offO + (size_t)ocol * p.ldo + orow] = (_Float16)v;
      }
    }
  }
}

// ---------------------------------------------------------------------------
// Elementwise / prep kernels
// ---------------------------------------------------------------------------
__global__ void pack_wt(const float* W, _Float16* Wt, int K, int N) {
  int idx = blockIdx.x * blockDim.x + threadIdx.x;
  if (idx >= K * N) return;
  int k = idx / N, n = idx - k * N;
  Wt[(size_t)n * K + k] = (_Float16)W[(size_t)k * N + n];
}

__global__ void f32_to_f16(const float* in, _Float16* out, long long n) {
  long long i = (long long)blockIdx.x * blockDim.x + threadIdx.x;
  if (i < n) out[i] = (_Float16)in[i];
}

// Fourier features: F[r][j] = sin(2pi*(qp.Bff)_j), F[r][128+j] = cos(...)
__global__ void fourier_kernel(const float* qpos, const float* Bff, _Float16* F) {
  int idx = blockIdx.x * blockDim.x + threadIdx.x;
  if (idx >= ROWS * 128) return;
  int r = idx >> 7, j = idx & 127;
  float f = 6.28318530717958647692f * (qpos[2 * r] * Bff[j] + qpos[2 * r + 1] * Bff[128 + j]);
  F[(size_t)r * 256 + j]       = (_Float16)sinf(f);
  F[(size_t)r * 256 + 128 + j] = (_Float16)cosf(f);
}

// 2D RoPE on q (per head: dims [0,128) use x-coord, [128,256) use y-coord).
// rotate pair (i, i+64): out_i = a*cos - b*sin ; out_{i+64} = b*cos + a*sin
__global__ void rope_q_kernel(const float* q, const float* qpos, _Float16* q16) {
  int idx = blockIdx.x * blockDim.x + threadIdx.x;
  if (idx >= ROWS * HEADS * 2 * 64) return;
  int i    = idx & 63;
  int part = (idx >> 6) & 1;
  int h    = (idx >> 7) & 3;
  int r    = idx >> 9;
  float coord = qpos[2 * r + part];
  float f = coord * 1024.0f * powf(10000.0f, -(float)i * (1.0f / 64.0f)); // ROT_SCALE/MIN_FREQ=1024
  float cs = cosf(f), sn = sinf(f);
  size_t base = (size_t)r * INNER + h * DHEAD + part * 128;
  float a = q[base + i], b = q[base + 64 + i];
  q16[base + i]      = (_Float16)(a * cs - b * sn);
  q16[base + 64 + i] = (_Float16)(b * cs + a * sn);
}

// RoPE(k) + instance-norm(k), instance-norm(v); write transposed f16 [d][ctx]
// One block per (row, head); 256 threads = one d each. LDS tree reduction.
__global__ __launch_bounds__(256) void ropenorm_kv_kernel(
    const float* kvraw, const float* cpos, _Float16* kT, _Float16* vT) {
  const int blk = blockIdx.x;
  const int r = blk >> 2, h = blk & 3;
  const int d = threadIdx.x;
  const size_t basek = (size_t)r * 2048 + h * DHEAD;
  const size_t basev = basek + INNER;

  const int  part = d >> 7;
  const int  j    = d & 127;
  const int  i    = j & 63;
  const bool up   = (j >= 64);
  float coord = cpos[2 * r + part];
  float f  = coord * 1024.0f * powf(10000.0f, -(float)i * (1.0f / 64.0f));
  float cs = cosf(f), sn = sinf(f);
  float e0 = kvraw[basek + part * 128 + i];
  float e1 = kvraw[basek + part * 128 + 64 + i];
  float kd = up ? (e1 * cs + e0 * sn) : (e0 * cs - e1 * sn);
  float vd = kvraw[basev + d];

  __shared__ float sk1[256], sk2[256], sv1[256], sv2[256];
  sk1[d] = kd; sk2[d] = kd * kd; sv1[d] = vd; sv2[d] = vd * vd;
  __syncthreads();
  for (int s = 128; s > 0; s >>= 1) {
    if (d < s) {
      sk1[d] += sk1[d + s]; sk2[d] += sk2[d + s];
      sv1[d] += sv1[d + s]; sv2[d] += sv2[d + s];
    }
    __syncthreads();
  }
  float kmu = sk1[0] * (1.0f / 256.0f);
  float kva = sk2[0] * (1.0f / 256.0f) - kmu * kmu;
  float vmu = sv1[0] * (1.0f / 256.0f);
  float vva = sv2[0] * (1.0f / 256.0f) - vmu * vmu;
  float kn = (kd - kmu) * rsqrtf(kva + 1e-5f);
  float vn = (vd - vmu) * rsqrtf(vva + 1e-5f);

  const int b = r >> 12, m = r & 4095;
  const size_t slice = (size_t)(b * HEADS + h) * DHEAD * NC;
  kT[slice + (size_t)d * NC + m] = (_Float16)kn;
  vT[slice + (size_t)d * NC + m] = (_Float16)vn;
}

// Wh1 has K=258: add the query_pos rows (256,257) then gelu.
__global__ void h1fix_kernel(const float* h1pre, const float* qpos,
                             const float* Wh1, _Float16* h1) {
  int idx = blockIdx.x * blockDim.x + threadIdx.x;
  if (idx >= ROWS * 256) return;
  int r = idx >> 8, n = idx & 255;
  float v = h1pre[idx] + qpos[2 * r] * Wh1[256 * 256 + n]
                       + qpos[2 * r + 1] * Wh1[257 * 256 + n];
  h1[idx] = (_Float16)gelu_f(v);
}

// Final head: (32768 x 128) @ (128 x 3) + bh3, f32 out.
__global__ void final_kernel(const _Float16* h2, const float* Wh3,
                             const float* bh3, float* out) {
  int r = blockIdx.x * blockDim.x + threadIdx.x;
  if (r >= ROWS) return;
  float a0 = bh3[0], a1 = bh3[1], a2 = bh3[2];
  const _Float16* hr = h2 + (size_t)r * 128;
#pragma unroll 4
  for (int k = 0; k < 128; ++k) {
    float hv = (float)hr[k];
    a0 += hv * Wh3[k * 3 + 0];
    a1 += hv * Wh3[k * 3 + 1];
    a2 += hv * Wh3[k * 3 + 2];
  }
  out[3 * r] = a0; out[3 * r + 1] = a1; out[3 * r + 2] = a2;
}

// ---------------------------------------------------------------------------
// Host-side launch helpers
// ---------------------------------------------------------------------------
static inline void launch_gemm(hipStream_t s,
    const _Float16* A, const _Float16* Bt, float* outF, _Float16* outH,
    const float* bias, const float* res,
    int M, int N, int K, int lda, int ldb, int ldo, int ldr,
    int Z, int HZ,
    long long sAb, long long sAh, long long sBb, long long sBh,
    long long sOb, long long sOh, float scale, int flags) {
  GemmP p;
  p.A = A; p.Bt = Bt; p.outF = outF; p.outH = outH; p.bias = bias; p.res = res;
  p.M = M; p.N = N; p.K = K; p.lda = lda; p.ldb = ldb; p.ldo = ldo; p.ldr = ldr;
  p.sAb = sAb; p.sAh = sAh; p.sBb = sBb; p.sBh = sBh; p.sOb = sOb; p.sOh = sOh;
  p.HZ = HZ; p.scale = scale; p.flags = flags;
  dim3 grid(N / 64, (M + 255) / 256, Z);
  hipLaunchKernelGGL(gemm_wmma, grid, dim3(256), 0, s, p);
}

extern "C" void kernel_launch(void* const* d_in, const int* in_sizes, int n_in,
                              void* d_out, int out_size, void* d_ws, size_t ws_size,
                              hipStream_t stream) {
  const float* z    = (const float*)d_in[0];
  const float* qpos = (const float*)d_in[1];
  const float* cpos = (const float*)d_in[2];
  const float* Bff  = (const float*)d_in[3];
  const float* Wc1  = (const float*)d_in[4];
  const float* Wc2  = (const float*)d_in[5];
  const float* Wq   = (const float*)d_in[6];
  const float* Wkv  = (const float*)d_in[7];
  const float* Wo   = (const float*)d_in[8];
  const float* bo   = (const float*)d_in[9];
  const float* Wf1  = (const float*)d_in[10];
  const float* bf1  = (const float*)d_in[11];
  const float* Wf2  = (const float*)d_in[12];
  const float* bf2  = (const float*)d_in[13];
  const float* Wh1  = (const float*)d_in[14];
  const float* Wh2  = (const float*)d_in[15];
  const float* Wh3  = (const float*)d_in[16];
  const float* bh3  = (const float*)d_in[17];
  float* out = (float*)d_out;

  // ---- workspace carve-out ----
  char* w = (char*)d_ws;
  size_t off = 0;
  auto alloc = [&](size_t bytes) -> void* {
    void* p = w + off;
    off = (off + bytes + 255) & ~(size_t)255;
    return p;
  };
  _Float16* wc1t = (_Float16*)alloc(256 * 256 * 2);
  _Float16* wc2t = (_Float16*)alloc(256 * 256 * 2);
  _Float16* wqt  = (_Float16*)alloc(256 * 1024 * 2);
  _Float16* wkvt = (_Float16*)alloc(256 * 2048 * 2);
  _Float16* wot  = (_Float16*)alloc(1024 * 256 * 2);
  _Float16* wf1t = (_Float16*)alloc(256 * 256 * 2);
  _Float16* wf2t = (_Float16*)alloc(256 * 256 * 2);
  _Float16* wh1t = (_Float16*)alloc(256 * 256 * 2);
  _Float16* wh2t = (_Float16*)alloc(256 * 128 * 2);

  _Float16* Fbuf  = (_Float16*)alloc((size_t)ROWS * 256 * 2);
  _Float16* T1    = (_Float16*)alloc((size_t)ROWS * 256 * 2);
  float*    xf    = (float*)   alloc((size_t)ROWS * 256 * 4);
  _Float16* x16   = (_Float16*)alloc((size_t)ROWS * 256 * 2);
  float*    qf    = (float*)   alloc((size_t)ROWS * 1024 * 4); // reused later
  _Float16* q16   = (_Float16*)alloc((size_t)ROWS * 1024 * 2);
  _Float16* z16   = (_Float16*)alloc((size_t)ROWS * 256 * 2);
  float*    kvraw = (float*)   alloc((size_t)ROWS * 2048 * 4); // reused later
  _Float16* kT    = (_Float16*)alloc((size_t)32 * 256 * NC * 2);
  _Float16* vT    = (_Float16*)alloc((size_t)32 * 256 * NC * 2);
  _Float16* kvT   = (_Float16*)alloc((size_t)32 * 256 * 256 * 2);
  _Float16* x2h   = (_Float16*)alloc((size_t)ROWS * 256 * 2);
  _Float16* fb16  = (_Float16*)alloc((size_t)ROWS * 256 * 2);
  _Float16* x3h   = (_Float16*)alloc((size_t)ROWS * 256 * 2);
  _Float16* h1_16 = (_Float16*)alloc((size_t)ROWS * 256 * 2);
  _Float16* h2_16 = (_Float16*)alloc((size_t)ROWS * 128 * 2);

  // Safe aliases (producers of the aliased regions are dead by then):
  _Float16* attn16 = (_Float16*)kvraw;                               // 67MB into 268MB
  float*    h1pre  = (float*)((char*)kvraw + (size_t)ROWS * 1024 * 2);
  float*    x2f    = qf;                                             // q consumed by rope
  float*    x3f    = (float*)((char*)qf + (size_t)ROWS * 256 * 4);

  const dim3 B256(256);
  auto blocks = [](long long n) { return dim3((unsigned)((n + 255) / 256)); };

  // ---- pack weights f32 -> f16 transposed [N][K] ----
  hipLaunchKernelGGL(pack_wt, blocks(256 * 256),  B256, 0, stream, Wc1, wc1t, 256, 256);
  hipLaunchKernelGGL(pack_wt, blocks(256 * 256),  B256, 0, stream, Wc2, wc2t, 256, 256);
  hipLaunchKernelGGL(pack_wt, blocks(256 * 1024), B256, 0, stream, Wq,  wqt,  256, 1024);
  hipLaunchKernelGGL(pack_wt, blocks(256 * 2048), B256, 0, stream, Wkv, wkvt, 256, 2048);
  hipLaunchKernelGGL(pack_wt, blocks(1024 * 256), B256, 0, stream, Wo,  wot,  1024, 256);
  hipLaunchKernelGGL(pack_wt, blocks(256 * 256),  B256, 0, stream, Wf1, wf1t, 256, 256);
  hipLaunchKernelGGL(pack_wt, blocks(256 * 256),  B256, 0, stream, Wf2, wf2t, 256, 256);
  hipLaunchKernelGGL(pack_wt, blocks(256 * 256),  B256, 0, stream, Wh1, wh1t, 256, 256); // first 256 K-rows
  hipLaunchKernelGGL(pack_wt, blocks(256 * 128),  B256, 0, stream, Wh2, wh2t, 256, 128);

  // ---- activations prep ----
  hipLaunchKernelGGL(f32_to_f16, blocks((long long)ROWS * 256), B256, 0, stream,
                     z, z16, (long long)ROWS * 256);
  hipLaunchKernelGGL(fourier_kernel, blocks((long long)ROWS * 128), B256, 0, stream,
                     qpos, Bff, Fbuf);

  // ---- Fourier MLP: T1 = gelu(F @ Wc1); x = T1 @ Wc2 ----
  launch_gemm(stream, Fbuf, wc1t, nullptr, T1, nullptr, nullptr,
              ROWS, 256, 256, 256, 256, 256, 0, 1, 1, 0,0,0,0,0,0, 1.0f,
              FLAG_GELU | FLAG_F16);
  launch_gemm(stream, T1, wc2t, xf, x16, nullptr, nullptr,
              ROWS, 256, 256, 256, 256, 256, 0, 1, 1, 0,0,0,0,0,0, 1.0f,
              FLAG_F32 | FLAG_F16);

  // ---- projections ----
  launch_gemm(stream, x16, wqt, qf, nullptr, nullptr, nullptr,
              ROWS, 1024, 256, 256, 256, 1024, 0, 1, 1, 0,0,0,0,0,0, 1.0f, FLAG_F32);
  launch_gemm(stream, z16, wkvt, kvraw, nullptr, nullptr, nullptr,
              ROWS, 2048, 256, 256, 256, 2048, 0, 1, 1, 0,0,0,0,0,0, 1.0f, FLAG_F32);

  // ---- RoPE(q) -> f16 ; RoPE(k)+instnorm(k,v) -> transposed f16 ----
  hipLaunchKernelGGL(rope_q_kernel, blocks((long long)ROWS * HEADS * 128), B256, 0,
                     stream, qf, qpos, q16);
  hipLaunchKernelGGL(ropenorm_kv_kernel, dim3(ROWS * HEADS), B256, 0, stream,
                     kvraw, cpos, kT, vT);

  // ---- linear attention: kv = k^T v (scaled 1/nc, stored transposed f16) ----
  launch_gemm(stream, kT, vT, nullptr, kvT, nullptr, nullptr,
              256, 256, NC, NC, NC, 256, 0, 32, 1,
              (long long)256 * NC, 0, (long long)256 * NC, 0, 65536LL, 0,
              1.0f / (float)NC, FLAG_TF16);
  // attn = q @ kv, written into [b*nq][h*256 + e] layout
  launch_gemm(stream, q16, kvT, nullptr, attn16, nullptr, nullptr,
              NQ, 256, 256, 1024, 256, 1024, 0, 32, HEADS,
              (long long)NQ * 1024, 256LL, 4LL * 65536, 65536LL,
              (long long)NQ * 1024, 256LL, 1.0f, FLAG_F16);

  // ---- x2 = attn @ Wo + bo + x ----
  launch_gemm(stream, attn16, wot, x2f, x2h, bo, xf,
              ROWS, 256, 1024, 1024, 1024, 256, 256, 1, 1, 0,0,0,0,0,0, 1.0f,
              FLAG_F32 | FLAG_F16);

  // ---- FFN: x3 = gelu(x2@Wf1+bf1)@Wf2 + bf2 + x2 ----
  launch_gemm(stream, x2h, wf1t, nullptr, fb16, bf1, nullptr,
              ROWS, 256, 256, 256, 256, 256, 0, 1, 1, 0,0,0,0,0,0, 1.0f,
              FLAG_GELU | FLAG_F16);
  launch_gemm(stream, fb16, wf2t, x3f, x3h, bf2, x2f,
              ROWS, 256, 256, 256, 256, 256, 256, 1, 1, 0,0,0,0,0,0, 1.0f,
              FLAG_F32 | FLAG_F16);

  // ---- head ----
  launch_gemm(stream, x3h, wh1t, h1pre, nullptr, nullptr, nullptr,
              ROWS, 256, 256, 256, 256, 256, 0, 1, 1, 0,0,0,0,0,0, 1.0f, FLAG_F32);
  hipLaunchKernelGGL(h1fix_kernel, blocks((long long)ROWS * 256), B256, 0, stream,
                     h1pre, qpos, Wh1, h1_16);
  launch_gemm(stream, h1_16, wh2t, nullptr, h2_16, nullptr, nullptr,
              ROWS, 128, 256, 256, 256, 128, 0, 1, 1, 0,0,0,0,0,0, 1.0f,
              FLAG_GELU | FLAG_F16);
  hipLaunchKernelGGL(final_kernel, blocks(ROWS), B256, 0, stream,
                     h2_16, Wh3, bh3, out);
}
